// GraphAttentionLayer_1357209666518
// MI455X (gfx1250) — matrix operand order repaired and checked by reading, not compiled, not run on hardware
//
#include <hip/hip_runtime.h>
#include <hip/hip_bf16.h>

typedef __attribute__((ext_vector_type(16))) _Float16 v16h;
typedef __attribute__((ext_vector_type(8)))  _Float16 v8h;
typedef __attribute__((ext_vector_type(8)))  float    v8f;
typedef __attribute__((ext_vector_type(4)))  float    v4f;
typedef __attribute__((ext_vector_type(4)))  int      v4i;

#define LRELU_ALPHA 0.2f
#define NEG_BIG    (-1.0e12f)

constexpr int B = 8, N = 2048, F = 128;

// ---- workspace layout (bytes, all regions 32B-aligned) ----
constexpr size_t OFF_U    = 0;                               // 256 f32  (u1|u2)
constexpr size_t OFF_S1   = 1024;                            // B*N f32
constexpr size_t OFF_S2   = OFF_S1 + (size_t)B*N*4;          // B*N f32
constexpr size_t OFF_MX   = OFF_S2 + (size_t)B*N*4;          // 8 f32 max, 8 f32 min
constexpr size_t OFF_WT   = OFF_MX + 256;                    // 128*256 f16 (wtrans^T)
constexpr size_t OFF_INPT = OFF_WT + (size_t)2*F*F*2;        // B*128*N f16 (inp^T)
constexpr size_t OFF_HD   = OFF_INPT + (size_t)B*F*N*2;      // B*N*256 f16 (h_double)

// u1[k] = sum_f W[k,f]*a[f];  u2[k] = sum_f W[k,f]*a[128+f]
__global__ void gat_prep_u_kernel(const float* __restrict__ W,
                                  const float* __restrict__ a,
                                  float* __restrict__ u) {
    const int k = threadIdx.x;                 // 128 threads
    float a1 = 0.f, a2 = 0.f;
    for (int f = 0; f < F; ++f) {
        const float w = W[k * F + f];
        a1 += w * a[f];
        a2 += w * a[F + f];
    }
    u[k]     = a1;
    u[F + k] = a2;
}

// s1[bn] = inp[bn,:]·u1 ;  s2[bn] = inp[bn,:]·u2
__global__ void gat_scores_kernel(const float* __restrict__ inp,
                                  const float* __restrict__ u,
                                  float* __restrict__ s1,
                                  float* __restrict__ s2) {
    const int idx = blockIdx.x * blockDim.x + threadIdx.x;   // B*N threads
    const float* x = inp + (size_t)idx * F;
    float a1 = 0.f, a2 = 0.f;
    for (int k = 0; k < F; ++k) {
        const float v = x[k];
        a1 += v * u[k];
        a2 += v * u[F + k];
    }
    s1[idx] = a1;
    s2[idx] = a2;
}

// per-batch max/min of s2 (for the softmax-stability bounds)
__global__ void gat_minmax_kernel(const float* __restrict__ s2,
                                  float* __restrict__ mx,
                                  float* __restrict__ mn) {
    __shared__ float smx[256], smn[256];
    const int b = blockIdx.x, t = threadIdx.x;
    float vmx = -3.4e38f, vmn = 3.4e38f;
    for (int j = t; j < N; j += 256) {
        const float v = s2[b * N + j];
        vmx = fmaxf(vmx, v);
        vmn = fminf(vmn, v);
    }
    smx[t] = vmx; smn[t] = vmn;
    __syncthreads();
    for (int s = 128; s > 0; s >>= 1) {
        if (t < s) {
            smx[t] = fmaxf(smx[t], smx[t + s]);
            smn[t] = fminf(smn[t], smn[t + s]);
        }
        __syncthreads();
    }
    if (t == 0) { mx[b] = smx[0]; mn[b] = smn[0]; }
}

// inpT[b,f,n] = (f16) inp[b,n,f]
__global__ void gat_transpose_kernel(const float* __restrict__ inp,
                                     _Float16* __restrict__ inpT) {
    const size_t idx = (size_t)blockIdx.x * 256 + threadIdx.x; // B*N*F threads
    const int f = (int)(idx & (F - 1));
    const size_t bn = idx >> 7;
    const int n = (int)(bn & (N - 1));
    const int b = (int)(bn >> 11);
    inpT[((size_t)b * F + f) * N + n] = (_Float16)inp[idx];
}

// wtT[c,k] = (f16) wtrans[k,c]
__global__ void gat_wtrans_kernel(const float* __restrict__ wtrans,
                                  _Float16* __restrict__ wtT) {
    const int idx = blockIdx.x * 256 + threadIdx.x;          // 2F*F threads
    const int c = idx & (F - 1);
    const int k = idx >> 7;
    wtT[(size_t)c * (2 * F) + k] = (_Float16)wtrans[idx];
}

// Main fused kernel: one wave per (b, 16-row tile).
// Single-pass masked double-softmax + f16 WMMA accumulation of
// attention@inp and (-neg_attention)@inp into h_double (f16).
__global__ __launch_bounds__(32) void gat_attn_kernel(
    const int* __restrict__ adj,
    const float* __restrict__ s1,
    const float* __restrict__ s2,
    const float* __restrict__ s2mx,
    const float* __restrict__ s2mn,
    const _Float16* __restrict__ inpT,
    _Float16* __restrict__ hdouble) {
    const int b    = blockIdx.y;
    const int i0   = blockIdx.x << 4;
    const int lane = threadIdx.x;
    const int l15  = lane & 15;   // A-matrix row / B,C column within tile
    const int half = lane >> 4;   // K-half selector

    const float s1r = s1[b * N + i0 + l15];
    // row-max upper bounds via monotone leaky-relu
    float tp = s1r + s2mx[b];
    const float mp = tp > 0.f ? tp : LRELU_ALPHA * tp;       // >= max masked e
    float tn = s1r + s2mn[b];
    const float mn = -(tn > 0.f ? tn : LRELU_ALPHA * tn);    // >= max masked -e

    const int*      adjrow = adj + ((size_t)b * N + (i0 + l15)) * (size_t)N;
    const float*    s2b    = s2 + b * N;
    const _Float16* inpTb  = inpT + (size_t)b * F * N;

    v8f accP[8] = {};
    v8f accN[8] = {};
    float zp0 = 0.f, zp1 = 0.f, zn0 = 0.f, zn1 = 0.f;

#pragma unroll 1
    for (int jb = 0; jb < N; jb += 32) {
        // each lane prefetches its own adj row 32 chunks ahead: one
        // global_prefetch covers the whole 16x32 chunk at jb+1024
        __builtin_prefetch(adjrow + jb + 1024, 0, 1);

        // this chunk's adj / s2 (latency hidden by the prefetch above)
        const int base0 = jb + (half << 4);                  // A K-run 0
        const int base1 = base0 + 8;                         // A K-run 1
        v4i A0 = *(const v4i*)(adjrow + base0);
        v4i A1 = *(const v4i*)(adjrow + base0 + 4);
        v4i A2 = *(const v4i*)(adjrow + base1);
        v4i A3 = *(const v4i*)(adjrow + base1 + 4);
        v4f S0 = *(const v4f*)(s2b + base0);
        v4f S1 = *(const v4f*)(s2b + base0 + 4);
        v4f S2 = *(const v4f*)(s2b + base1);
        v4f S3 = *(const v4f*)(s2b + base1 + 4);

        // B tiles group A (cols 0..63): latency covered by exp block below
        const int kb = jb + (half << 4);                     // B K-run
        v16h bvA[4];
#pragma unroll
        for (int c = 0; c < 4; ++c)
            bvA[c] = *(const v16h*)(inpTb + (size_t)((c << 4) + l15) * N + kb);

        int   av[16];
        float sv[16];
#pragma unroll
        for (int q = 0; q < 4; ++q) {
            av[q] = A0[q]; av[4 + q] = A1[q]; av[8 + q] = A2[q]; av[12 + q] = A3[q];
            sv[q] = S0[q]; sv[4 + q] = S1[q]; sv[8 + q] = S2[q]; sv[12 + q] = S3[q];
        }

        v16h ap, an;
#pragma unroll
        for (int t = 0; t < 16; ++t) {
            float e = s1r + sv[t];
            e = e > 0.f ? e : LRELU_ALPHA * e;               // leaky-relu
            const bool msk = av[t] > 0;
            const float pp = __expf((msk ?  e : NEG_BIG) - mp);
            const float pn = __expf((msk ? -e : NEG_BIG) - mn);
            if (t & 1) { zp1 += pp; zn1 += pn; }
            else       { zp0 += pp; zn0 += pn; }
            ap[t] = (_Float16)pp;
            an[t] = (_Float16)pn;
        }

        // B tiles group B (cols 64..127): latency covered by group-A WMMAs
        v16h bvB[4];
#pragma unroll
        for (int c = 0; c < 4; ++c)
            bvB[c] = *(const v16h*)(inpTb + (size_t)(((c + 4) << 4) + l15) * N + kb);

#pragma unroll
        for (int c = 0; c < 4; ++c) {
            accP[c] = __builtin_amdgcn_wmma_f32_16x16x32_f16(
                false, ap, false, bvA[c], (short)0, accP[c], false, false);
            accN[c] = __builtin_amdgcn_wmma_f32_16x16x32_f16(
                false, an, false, bvA[c], (short)0, accN[c], false, false);
        }
#pragma unroll
        for (int c = 0; c < 4; ++c) {
            accP[c + 4] = __builtin_amdgcn_wmma_f32_16x16x32_f16(
                false, ap, false, bvB[c], (short)0, accP[c + 4], false, false);
            accN[c + 4] = __builtin_amdgcn_wmma_f32_16x16x32_f16(
                false, an, false, bvB[c], (short)0, accN[c + 4], false, false);
        }
    }

    float zp = zp0 + zp1;
    float zn = zn0 + zn1;
    // full row sums: lane l holds partial for row l&15; partner is l^16
    zp += __shfl_xor(zp, 16, 32);
    zn += __shfl_xor(zn, 16, 32);

    _Float16* hdb = hdouble + (size_t)b * N * (2 * F);
#pragma unroll
    for (int r = 0; r < 8; ++r) {
        const int mrow = r + (half << 3);                // C-layout row
        const float izp =  1.0f / __shfl(zp, mrow, 32);
        const float izn = -1.0f / __shfl(zn, mrow, 32);  // negative attention sign
        _Float16* orow = hdb + (size_t)(i0 + mrow) * (2 * F);
#pragma unroll
        for (int c = 0; c < 8; ++c) {
            orow[(c << 4) + l15]     = (_Float16)(accP[c][r] * izp);
            orow[F + (c << 4) + l15] = (_Float16)(accN[c][r] * izn);
        }
    }
}

// Final GEMM: out = elu(h_double @ wtrans), one wave per 16-row tile, K=256.
__global__ __launch_bounds__(32) void gat_out_kernel(
    const _Float16* __restrict__ hdouble,   // [B*N, 256]
    const _Float16* __restrict__ wtT,       // [128, 256]
    float* __restrict__ out) {              // [B*N, 128]
    const int r0   = blockIdx.x << 4;
    const int lane = threadIdx.x;
    const int l15  = lane & 15;
    const int half = lane >> 4;
    const _Float16* arow = hdouble + (size_t)(r0 + l15) * (2 * F);

    v8f acc[8] = {};
#pragma unroll
    for (int kb = 0; kb < 2 * F; kb += 32) {
        v8h alo = *(const v8h*)(arow + kb + (half << 4));
        v8h ahi = *(const v8h*)(arow + kb + 8 + (half << 4));
        const int kk = kb + (half << 4);
        v16h av;
#pragma unroll
        for (int t = 0; t < 8; ++t) { av[t] = alo[t]; av[8 + t] = ahi[t]; }
#pragma unroll
        for (int c = 0; c < 8; ++c) {
            v16h bv = *(const v16h*)(wtT + (size_t)((c << 4) + l15) * (2 * F) + kk);
            acc[c] = __builtin_amdgcn_wmma_f32_16x16x32_f16(
                false, av, false, bv, (short)0, acc[c], false, false);
        }
    }
#pragma unroll
    for (int r = 0; r < 8; ++r) {
        const int mrow = r + (half << 3);
        float* orow = out + (size_t)(r0 + mrow) * F;
#pragma unroll
        for (int c = 0; c < 8; ++c) {
            const float x = acc[c][r];
            orow[(c << 4) + l15] = x > 0.f ? x : expm1f(x);  // elu
        }
    }
}

extern "C" void kernel_launch(void* const* d_in, const int* in_sizes, int n_in,
                              void* d_out, int out_size, void* d_ws, size_t ws_size,
                              hipStream_t stream) {
    const float* inp    = (const float*)d_in[0];   // [B,N,128] f32
    const int*   adj    = (const int*)  d_in[1];   // [B,N,N]   i32
    const float* W      = (const float*)d_in[2];   // [128,128] f32
    const float* a      = (const float*)d_in[3];   // [256,1]   f32
    const float* wtrans = (const float*)d_in[4];   // [256,128] f32

    char* ws = (char*)d_ws;
    float*    u    = (float*)(ws + OFF_U);
    float*    s1   = (float*)(ws + OFF_S1);
    float*    s2   = (float*)(ws + OFF_S2);
    float*    s2mx = (float*)(ws + OFF_MX);
    float*    s2mn = s2mx + 8;
    _Float16* wtT  = (_Float16*)(ws + OFF_WT);
    _Float16* inpT = (_Float16*)(ws + OFF_INPT);
    _Float16* hd   = (_Float16*)(ws + OFF_HD);

    gat_prep_u_kernel<<<1, 128, 0, stream>>>(W, a, u);
    gat_scores_kernel<<<(B * N) / 256, 256, 0, stream>>>(inp, u, s1, s2);
    gat_minmax_kernel<<<B, 256, 0, stream>>>(s2, s2mx, s2mn);
    gat_transpose_kernel<<<(B * N * F) / 256, 256, 0, stream>>>(inp, inpT);
    gat_wtrans_kernel<<<(2 * F * F) / 256, 256, 0, stream>>>(wtrans, wtT);

    dim3 g4(N / 16, B);
    gat_attn_kernel<<<g4, 32, 0, stream>>>(adj, s1, s2, s2mx, s2mn, inpT, hd);
    gat_out_kernel<<<(B * N) / 16, 32, 0, stream>>>(hd, wtT, (float*)d_out);
}
